// ParallelFFNMoE_32804960207216
// MI455X (gfx1250) — matrix operand order repaired and checked by reading, not compiled, not run on hardware
//
#include <hip/hip_runtime.h>

// ---------------------------------------------------------------------------
// ParallelFFNMoE for MI455X (gfx1250, wave32, WMMA).
// Compute-bound (~412 GFLOP vs ~15us of HBM traffic) -> run all GEMMs on the
// bf16 WMMA path with a hi/lo bf16 split (bf16x3) for ~fp32 accuracy:
//   A*B ~= Ahi*Bhi + Ahi*Blo + Alo*Bhi   (f32 accumulate, 3 wmma / tile-step)
// v2: ping-pong LDS double buffering (global loads overlap wmma) + hardware
//     V_TANH_F32 for the GELU epilogue.
// ---------------------------------------------------------------------------

typedef __attribute__((ext_vector_type(16))) __bf16 v16bf;
typedef __attribute__((ext_vector_type(8)))  float  v8f;

#define BM 128
#define BN 64
#define BK 32
#define APAD 36   // LDS row stride (halfs) for A tiles: 72B rows, 8B-aligned chunks
#define BPAD 36   // LDS row stride for B^T tiles

__device__ __forceinline__ unsigned short bf16_rne(float f) {
  unsigned int u = __float_as_uint(f);
  u += 0x7FFFu + ((u >> 16) & 1u);           // round-to-nearest-even
  return (unsigned short)(u >> 16);
}
__device__ __forceinline__ float bf16_f32(unsigned short h) {
  return __uint_as_float(((unsigned int)h) << 16);
}

__device__ __forceinline__ float fast_tanh(float u) {
#if __has_builtin(__builtin_amdgcn_tanhf)
  return __builtin_amdgcn_tanhf(u);          // V_TANH_F32 (gfx1250 TRANS op)
#elif __has_builtin(__builtin_amdgcn_tanh_f32)
  return __builtin_amdgcn_tanh_f32(u);
#else
  return tanhf(u);
#endif
}
__device__ __forceinline__ float gelu_tanh(float v) {
  const float c = 0.7978845608028654f;       // sqrt(2/pi)
  float t = fast_tanh(c * (v + 0.044715f * v * v * v));
  return 0.5f * v * (1.0f + t);
}

union Frag {
  v16bf v;
  uint2 q[4];
};

// C[M,N] = epilogue( A[M,K] @ B[K,N] + bias[N] )
// EPI 0: gelu -> store f32      (FFN hidden / expert hidden)
// EPI 1: store f32              (ffn_out, initializes d_out)
// EPI 2: out += LAM*mask*gate*v (per-expert MoE accumulate)
template <int EPI>
__global__ __launch_bounds__(256)
void gemm_bf16x3(const float* __restrict__ A, const float* __restrict__ Bm,
                 const float* __restrict__ bias, float* __restrict__ C,
                 int K, int N, const float* __restrict__ gates, int expert,
                 const int* __restrict__ idp, int Sdim)
{
  __shared__ unsigned short Ah[2][BM][APAD];
  __shared__ unsigned short Al[2][BM][APAD];
  __shared__ unsigned short Bh[2][BN][BPAD];
  __shared__ unsigned short Bl[2][BN][BPAD];

  const int tid  = threadIdx.x;
  const int m0   = blockIdx.y * BM;
  const int n0   = blockIdx.x * BN;

  const int wave = tid >> 5;
  const int lane = tid & 31;
  const int hf   = lane >> 4;      // lane group (0: lanes 0-15, 1: lanes 16-31)
  const int l15  = lane & 15;
  const int wm   = wave >> 1;      // 0..3 : 32-row slab
  const int wn   = wave & 1;       // 0..1 : 32-col slab

  // per-thread global staging coordinates
  const int arow = tid >> 3;       // + 32 per i   (8 float4 per 32-float A row)
  const int ac4  = tid & 7;
  const int brow = tid >> 4;       // + 16 per i   (16 float4 per 64-float B row)
  const int bc4  = tid & 15;
  const float* abase = A + (size_t)(m0 + arow) * K + ac4 * 4;
  const float* bbase = Bm + (size_t)brow * N + n0 + bc4 * 4;

  v8f acc[2][2];
  {
    v8f z = {0.f, 0.f, 0.f, 0.f, 0.f, 0.f, 0.f, 0.f};
    acc[0][0] = z; acc[0][1] = z; acc[1][0] = z; acc[1][1] = z;
  }

  // CDNA5 bf16 fragment layout (ISA 7.12.2):
  //  A 16x32: lanes 0-15 hold K 0..7 & 16..23 of row M=lane; lanes 16-31 hold
  //           K 8..15 & 24..31 of row M=lane-16.
  //  B 32x16: lanes 0-15 hold K 0..7 & 16..23 of col N=lane; lanes 16-31 hold
  //           K 8..15 & 24..31 of col N=lane-16.
  const int kb0 = hf ? 8 : 0;
  const int kbB = hf ? 8 : 0;

  float4 areg[4];
  float4 breg[2];

  auto load_tile = [&](int k0) {
    #pragma unroll
    for (int i = 0; i < 4; ++i)
      areg[i] = *(const float4*)(abase + (size_t)i * 32 * K + k0);
    #pragma unroll
    for (int i = 0; i < 2; ++i)
      breg[i] = *(const float4*)(bbase + (size_t)(k0 + i * 16) * N);
  };

  auto store_tile = [&](int buf) {
    #pragma unroll
    for (int i = 0; i < 4; ++i) {
      const int row = arow + i * 32;
      const float4 val = areg[i];
      unsigned short h0 = bf16_rne(val.x), h1 = bf16_rne(val.y);
      unsigned short h2 = bf16_rne(val.z), h3 = bf16_rne(val.w);
      unsigned short g0 = bf16_rne(val.x - bf16_f32(h0));
      unsigned short g1 = bf16_rne(val.y - bf16_f32(h1));
      unsigned short g2 = bf16_rne(val.z - bf16_f32(h2));
      unsigned short g3 = bf16_rne(val.w - bf16_f32(h3));
      *(uint2*)&Ah[buf][row][ac4 * 4] =
          make_uint2((unsigned)h0 | ((unsigned)h1 << 16), (unsigned)h2 | ((unsigned)h3 << 16));
      *(uint2*)&Al[buf][row][ac4 * 4] =
          make_uint2((unsigned)g0 | ((unsigned)g1 << 16), (unsigned)g2 | ((unsigned)g3 << 16));
    }
    #pragma unroll
    for (int i = 0; i < 2; ++i) {
      const int row = brow + i * 16;
      const float4 val = breg[i];
      float vv[4] = {val.x, val.y, val.z, val.w};
      #pragma unroll
      for (int j = 0; j < 4; ++j) {
        unsigned short hh = bf16_rne(vv[j]);
        Bh[buf][bc4 * 4 + j][row] = hh;
        Bl[buf][bc4 * 4 + j][row] = bf16_rne(vv[j] - bf16_f32(hh));
      }
    }
  };

  // prologue: stage tile 0
  load_tile(0);
  store_tile(0);
  __syncthreads();

  const int nk = K / BK;
  for (int kt = 0; kt < nk; ++kt) {
    const int cur = kt & 1;

    // issue next tile's global loads before the wmma block (latency overlap)
    if (kt + 1 < nk) load_tile((kt + 1) * BK);

    // ---- A fragments for this wave's two 16-row tiles ----
    Frag ahf[2], alf[2];
    #pragma unroll
    for (int mi = 0; mi < 2; ++mi) {
      const unsigned short* pr = &Ah[cur][wm * 32 + mi * 16 + l15][0];
      ahf[mi].q[0] = *(const uint2*)(pr + kb0);
      ahf[mi].q[1] = *(const uint2*)(pr + kb0 + 4);
      ahf[mi].q[2] = *(const uint2*)(pr + kb0 + 16);
      ahf[mi].q[3] = *(const uint2*)(pr + kb0 + 20);
      const unsigned short* pl = &Al[cur][wm * 32 + mi * 16 + l15][0];
      alf[mi].q[0] = *(const uint2*)(pl + kb0);
      alf[mi].q[1] = *(const uint2*)(pl + kb0 + 4);
      alf[mi].q[2] = *(const uint2*)(pl + kb0 + 16);
      alf[mi].q[3] = *(const uint2*)(pl + kb0 + 20);
    }

    // ---- B fragments + 3 wmma per output tile (bf16x3) ----
    #pragma unroll
    for (int ni = 0; ni < 2; ++ni) {
      Frag bhf, blf;
      const unsigned short* pr = &Bh[cur][wn * 32 + ni * 16 + l15][0];
      bhf.q[0] = *(const uint2*)(pr + kbB);
      bhf.q[1] = *(const uint2*)(pr + kbB + 4);
      bhf.q[2] = *(const uint2*)(pr + kbB + 16);
      bhf.q[3] = *(const uint2*)(pr + kbB + 20);
      const unsigned short* pl = &Bl[cur][wn * 32 + ni * 16 + l15][0];
      blf.q[0] = *(const uint2*)(pl + kbB);
      blf.q[1] = *(const uint2*)(pl + kbB + 4);
      blf.q[2] = *(const uint2*)(pl + kbB + 16);
      blf.q[3] = *(const uint2*)(pl + kbB + 20);
      #pragma unroll
      for (int mi = 0; mi < 2; ++mi) {
        acc[mi][ni] = __builtin_amdgcn_wmma_f32_16x16x32_bf16(
            false, ahf[mi].v, false, bhf.v, (short)0, acc[mi][ni], false, false);
        acc[mi][ni] = __builtin_amdgcn_wmma_f32_16x16x32_bf16(
            false, ahf[mi].v, false, blf.v, (short)0, acc[mi][ni], false, false);
        acc[mi][ni] = __builtin_amdgcn_wmma_f32_16x16x32_bf16(
            false, alf[mi].v, false, bhf.v, (short)0, acc[mi][ni], false, false);
      }
    }

    // stage next tile into the other buffer; reads of `cur` finished above for
    // this wave, and `cur^1` was last read before the previous barrier.
    if (kt + 1 < nk) {
      store_tile(cur ^ 1);
      __syncthreads();
    }
  }

  // ---- epilogue ----
  // C/D layout: VGPR r, lanes 0-15 -> (M=r, N=lane); lanes 16-31 -> (M=8+r, N=lane-16)
  const int id_cut = (EPI == 2) ? *idp : 0;
  #pragma unroll
  for (int mi = 0; mi < 2; ++mi) {
    #pragma unroll
    for (int ni = 0; ni < 2; ++ni) {
      const int col = n0 + wn * 32 + ni * 16 + l15;
      const float bb = bias[col];
      #pragma unroll
      for (int r = 0; r < 8; ++r) {
        const int row = m0 + wm * 32 + mi * 16 + hf * 8 + r;
        const float v = acc[mi][ni][r] + bb;
        const size_t off = (size_t)row * N + col;
        if (EPI == 0) {
          C[off] = gelu_tanh(v);
        } else if (EPI == 1) {
          C[off] = v;
        } else {
          const int s = row % Sdim;                // token position in sequence
          if (s >= id_cut) {
            const float g = gates[row * 8 + expert];
            C[off] += 0.5f * g * v;                // LAM = 0.5
          }
        }
      }
    }
  }
}

// ---------------------------------------------------------------------------
// Router: one wave per token. logits = x_row @ Wg (D x 8), then top-2 of the
// softmax renormalized (softmax denominator cancels: g = e_i / (e_1 + e_2)).
// ---------------------------------------------------------------------------
__global__ __launch_bounds__(256)
void router_kernel(const float* __restrict__ x, const float* __restrict__ Wg,
                   float* __restrict__ gates, int D)
{
  const int token = blockIdx.x * 8 + (threadIdx.x >> 5);
  const int lane  = threadIdx.x & 31;

  float a0 = 0.f, a1 = 0.f, a2 = 0.f, a3 = 0.f;
  float a4 = 0.f, a5 = 0.f, a6 = 0.f, a7 = 0.f;
  const float* xr = x + (size_t)token * D;
  for (int i = lane; i < D; i += 32) {
    const float xv = xr[i];
    const float4 w0 = *(const float4*)(Wg + (size_t)i * 8);
    const float4 w1 = *(const float4*)(Wg + (size_t)i * 8 + 4);
    a0 += xv * w0.x; a1 += xv * w0.y; a2 += xv * w0.z; a3 += xv * w0.w;
    a4 += xv * w1.x; a5 += xv * w1.y; a6 += xv * w1.z; a7 += xv * w1.w;
  }
  #pragma unroll
  for (int off = 16; off > 0; off >>= 1) {
    a0 += __shfl_xor(a0, off, 32); a1 += __shfl_xor(a1, off, 32);
    a2 += __shfl_xor(a2, off, 32); a3 += __shfl_xor(a3, off, 32);
    a4 += __shfl_xor(a4, off, 32); a5 += __shfl_xor(a5, off, 32);
    a6 += __shfl_xor(a6, off, 32); a7 += __shfl_xor(a7, off, 32);
  }
  if (lane == 0) {
    float l[8] = {a0, a1, a2, a3, a4, a5, a6, a7};
    float m = l[0];
    #pragma unroll
    for (int e = 1; e < 8; ++e) m = fmaxf(m, l[e]);
    float p[8];
    #pragma unroll
    for (int e = 0; e < 8; ++e) p[e] = __expf(l[e] - m);
    int i1 = 0; float v1 = p[0];
    #pragma unroll
    for (int e = 1; e < 8; ++e) if (p[e] > v1) { v1 = p[e]; i1 = e; }
    int i2 = -1; float v2 = -1.f;
    #pragma unroll
    for (int e = 0; e < 8; ++e) if (e != i1 && p[e] > v2) { v2 = p[e]; i2 = e; }
    const float inv = 1.0f / (v1 + v2);
    #pragma unroll
    for (int e = 0; e < 8; ++e)
      gates[(size_t)token * 8 + e] = (e == i1) ? v1 * inv : ((e == i2) ? v2 * inv : 0.f);
  }
}

// ---------------------------------------------------------------------------
extern "C" void kernel_launch(void* const* d_in, const int* in_sizes, int n_in,
                              void* d_out, int out_size, void* d_ws, size_t ws_size,
                              hipStream_t stream)
{
  const float* x   = (const float*)d_in[0];
  const float* W1  = (const float*)d_in[1];
  const float* b1  = (const float*)d_in[2];
  const float* W2  = (const float*)d_in[3];
  const float* b2  = (const float*)d_in[4];
  const float* Wg  = (const float*)d_in[5];
  const float* We1 = (const float*)d_in[6];
  const float* be1 = (const float*)d_in[7];
  const float* We2 = (const float*)d_in[8];
  const float* be2 = (const float*)d_in[9];
  const int*   idp = (const int*)d_in[10];
  float* out = (float*)d_out;

  const int Bb = 4, S = 2048, D = 1024, DFF = 4096, E = 8, DFE = 1024;
  const int T = Bb * S;                        // 8192 tokens

  // ws layout: H1 f32 [T,DFF] (128MB) | h f32 [T,DFE] (32MB) | gates [T,E]
  char* ws = (char*)d_ws;
  float* H1    = (float*)ws;
  float* h     = (float*)(ws + (size_t)T * DFF * 4);
  float* gates = (float*)(ws + (size_t)T * DFF * 4 + (size_t)T * DFE * 4);

  // 1) router gates
  router_kernel<<<T / 8, 256, 0, stream>>>(x, Wg, gates, D);

  // 2) dense FFN: H1 = gelu(x@W1 + b1); out = H1@W2 + b2
  gemm_bf16x3<0><<<dim3(DFF / BN, T / BM), 256, 0, stream>>>(
      x, W1, b1, H1, D, DFF, nullptr, 0, idp, S);
  gemm_bf16x3<1><<<dim3(D / BN, T / BM), 256, 0, stream>>>(
      H1, W2, b2, out, DFF, D, nullptr, 0, idp, S);

  // 3) MoE: per expert, h = gelu(x@We1[e]+be1[e]); out += LAM*mask*gate*(h@We2[e]+be2[e])
  for (int e = 0; e < E; ++e) {
    gemm_bf16x3<0><<<dim3(DFE / BN, T / BM), 256, 0, stream>>>(
        x, We1 + (size_t)e * D * DFE, be1 + (size_t)e * DFE, h, D, DFE,
        nullptr, 0, idp, S);
    gemm_bf16x3<2><<<dim3(D / BN, T / BM), 256, 0, stream>>>(
        h, We2 + (size_t)e * DFE * D, be2 + (size_t)e * D, out, DFE, D,
        gates, e, idp, S);
  }
}
